// HumanVAttention_15659450761362
// MI455X (gfx1250) — matrix operand order repaired and checked by reading, not compiled, not run on hardware
//
#include <hip/hip_runtime.h>
#include <hip/hip_bf16.h>
#include <stdint.h>

// ---------------------------------------------------------------------------
// Problem constants (from reference)
// ---------------------------------------------------------------------------
#define S_LEN   4096
#define HIDDEN  2048
#define NH      16
#define NKV     4
#define HD      128
#define BLK     128
#define NBLK    (S_LEN / BLK)     // 32
#define LOCAL   4
#define GLOB    4
#define KSTRIDE 8
#define NEGINF  (-1000000000.0f)
#define SCALING 0.08838834764831845f   // 128^-0.5

typedef __attribute__((ext_vector_type(16))) __bf16 v16bf;
typedef __attribute__((ext_vector_type(8)))  __bf16 v8bf;
typedef __attribute__((ext_vector_type(8)))  float  v8f;
typedef __attribute__((ext_vector_type(4)))  unsigned int v4u;
typedef __attribute__((ext_vector_type(8)))  int v8i;
typedef __attribute__((ext_vector_type(4)))  int v4i;

// ---------------------------------------------------------------------------
// Helpers
// ---------------------------------------------------------------------------
__device__ __forceinline__ float bf2f(__bf16 b) {
    union { unsigned short u; __bf16 b; } in; in.b = b;
    union { unsigned u; float f; } out; out.u = ((unsigned)in.u) << 16;
    return out.f;
}
__device__ __forceinline__ __bf16 f2bf(float f) {
    union { float f; unsigned u; } in; in.f = f;
    unsigned r = (in.u + 0x7FFFu + ((in.u >> 16) & 1u)) >> 16;
    union { unsigned short u; __bf16 b; } out; out.u = (unsigned short)r;
    return out.b;
}
__device__ __forceinline__ v16bf cat8(v8bf a, v8bf b) {
    return __builtin_shufflevector(a, b, 0,1,2,3,4,5,6,7,8,9,10,11,12,13,14,15);
}
__device__ __forceinline__ v8f wmma_bf16(v16bf a, v16bf b, v8f c) {
    // D = A(16x32 bf16) * B(32x16 bf16) + C(16x16 f32)
    return __builtin_amdgcn_wmma_f32_16x16x32_bf16(false, a, false, b,
                                                   (short)0, c, false, false);
}

// ---------------------------------------------------------------------------
// Tensor Data Mover: 2-D tile (bf16) global -> LDS.
// D# bitfields per CDNA5 ISA.
//   group0: count=1 | lds_addr | global_addr[56:0] | type=2
//   group1: data_size=1(2B); tensor_dim0/1; tile_dim0/1; tensor_dim0_stride
// Remaining groups zero (<=2D tensor). 6-arg clang-23 builtin form.
// ---------------------------------------------------------------------------
__device__ __forceinline__ void tdm_load_2d_bf16(uint32_t lds_off, const void* gaddr,
                                                 uint32_t tensor_d0, uint32_t tensor_d1,
                                                 uint32_t stride0_elems,
                                                 uint32_t tile_d0, uint32_t tile_d1) {
    uint64_t ga = (uint64_t)(uintptr_t)gaddr;
    v4u g0;
    g0[0] = 1u;                                              // count=1, user-mode
    g0[1] = lds_off;                                         // LDS byte address
    g0[2] = (uint32_t)(ga & 0xFFFFFFFFu);                    // global_addr[31:0]
    g0[3] = (uint32_t)((ga >> 32) & 0x01FFFFFFu) | (2u << 30); // [56:32] | type=2
    v8i g1;
    g1[0] = (int)(1u << 16);                                 // data_size=1 -> 2 bytes
    g1[1] = (int)((tensor_d0 & 0xFFFFu) << 16);              // tensor_dim0[15:0]
    g1[2] = (int)(((tensor_d0 >> 16) & 0xFFFFu) |
                  ((tensor_d1 & 0xFFFFu) << 16));            // dim0[31:16] | dim1[15:0]
    g1[3] = (int)(((tensor_d1 >> 16) & 0xFFFFu) |
                  ((tile_d0 & 0xFFFFu) << 16));              // dim1[31:16] | tile_dim0
    g1[4] = (int)(tile_d1 & 0xFFFFu);                        // tile_dim1, tile_dim2=0
    g1[5] = (int)stride0_elems;                              // tensor_dim0_stride[31:0]
    g1[6] = 0;                                               // stride0[47:32], dim1_stride lo
    g1[7] = 0;
    v4i z4 = {0, 0, 0, 0};
    v8i z8 = {0, 0, 0, 0, 0, 0, 0, 0};
    __builtin_amdgcn_tensor_load_to_lds(g0, g1, z4, z4, z8, 0);
}

// ---------------------------------------------------------------------------
// fp32 -> bf16 conversion
// ---------------------------------------------------------------------------
__global__ void cvt_f32_bf16(const float* __restrict__ src,
                             __bf16* __restrict__ dst, int n) {
    int i = blockIdx.x * blockDim.x + threadIdx.x;
    if (i < n) dst[i] = f2bf(src[i]);
}

// ---------------------------------------------------------------------------
// RoPE in-place on bf16 activations laid out [S][ncols], heads of 128 cols.
// ---------------------------------------------------------------------------
__global__ void rope_kernel(__bf16* __restrict__ X,
                            const float* __restrict__ cosb,
                            const float* __restrict__ sinb,
                            int ncols, int total_pairs) {
    int p = blockIdx.x * blockDim.x + threadIdx.x;
    if (p >= total_pairs) return;
    int pairs_per_row = ncols >> 1;
    int s    = p / pairs_per_row;
    int rem  = p - s * pairs_per_row;
    int head = rem >> 6;
    int d    = rem & 63;
    size_t base = (size_t)s * ncols + head * HD;
    float x1 = bf2f(X[base + d]);
    float x2 = bf2f(X[base + d + 64]);
    float c1 = cosb[(size_t)s * HD + d];
    float s1 = sinb[(size_t)s * HD + d];
    float c2 = cosb[(size_t)s * HD + d + 64];
    float s2 = sinb[(size_t)s * HD + d + 64];
    X[base + d]      = f2bf(x1 * c1 - x2 * s1);
    X[base + d + 64] = f2bf(x2 * c2 + x1 * s2);
}

// ---------------------------------------------------------------------------
// Generic bf16 GEMM: C[M,N] = A[M,K] * W[K,N]. 128x128 tile per 256-thread
// block (8 waves). W tile transposed through LDS.
// ---------------------------------------------------------------------------
template <bool BF16OUT>
__global__ __launch_bounds__(256)
void gemm_bf16_kernel(const __bf16* __restrict__ A, const __bf16* __restrict__ W,
                      void* __restrict__ Cout, int M, int N, int Kd) {
    __shared__ __bf16 Wt[128 * 32];   // Wt[col][k]

    const int tid  = threadIdx.x;
    const int wave = tid >> 5, lane = tid & 31;
    const int half = lane >> 4, lm = lane & 15;
    const int c0 = blockIdx.x * 128;
    const int m0 = blockIdx.y * 128 + wave * 16;

    const v8f zero = {0.f,0.f,0.f,0.f,0.f,0.f,0.f,0.f};
    v8f acc[8];
#pragma unroll
    for (int i = 0; i < 8; ++i) acc[i] = zero;

    const int kk = tid >> 3;          // 0..31
    const int cc = (tid & 7) << 4;    // 0..112

    const __bf16* arow = A + (size_t)(m0 + lm) * Kd;

    for (int k0 = 0; k0 < Kd; k0 += 32) {
        const __bf16* wsrc = W + (size_t)(k0 + kk) * N + c0 + cc;
        v8bf w0 = *(const v8bf*)(wsrc);
        v8bf w1 = *(const v8bf*)(wsrc + 8);
        if (k0 + 32 < Kd)                       // prefetch next W tile
            __builtin_prefetch(wsrc + (size_t)32 * N, 0, 1);
        __syncthreads();
#pragma unroll
        for (int j = 0; j < 8; ++j) Wt[(cc + j) * 32 + kk] = w0[j];
#pragma unroll
        for (int j = 0; j < 8; ++j) Wt[(cc + 8 + j) * 32 + kk] = w1[j];
        __syncthreads();

        v8bf alo = *(const v8bf*)(arow + k0 + half * 8);
        v8bf ahi = *(const v8bf*)(arow + k0 + 16 + half * 8);
        v16bf av = cat8(alo, ahi);

#pragma unroll
        for (int nt = 0; nt < 8; ++nt) {
            const __bf16* bp = &Wt[(nt * 16 + lm) * 32];
            v8bf blo = *(const v8bf*)(bp + half * 8);
            v8bf bhi = *(const v8bf*)(bp + 16 + half * 8);
            acc[nt] = wmma_bf16(av, cat8(blo, bhi), acc[nt]);
        }
    }

#pragma unroll
    for (int nt = 0; nt < 8; ++nt) {
#pragma unroll
        for (int r = 0; r < 8; ++r) {
            int row = m0 + r + half * 8;
            int col = c0 + nt * 16 + lm;
            float v = acc[nt][r];
            if (BF16OUT) ((__bf16*)Cout)[(size_t)row * N + col] = f2bf(v);
            else         ((float*)Cout)[(size_t)row * N + col] = v;
        }
    }
}

// ---------------------------------------------------------------------------
// Block-sparse attention. Grid (NBLK, NH), 256 threads (8 waves).
// K tile is DMA'd into LDS by the Tensor Data Mover each key block; V tile is
// staged transposed; flash-style online softmax; all matmuls via WMMA bf16.
// LDS: 32K (Kraw) + 32K (Vt) + 32K (Pt) + mask = ~96.5 KB (<320 KB/WGP).
// ---------------------------------------------------------------------------
__global__ __launch_bounds__(256)
void attn_kernel(const __bf16* __restrict__ Q, const __bf16* __restrict__ Kc,
                 const __bf16* __restrict__ Vc, const float* __restrict__ mask2d,
                 __bf16* __restrict__ O) {
    __shared__ __bf16 Kraw[BLK * HD];      // K tile [key][hd] (TDM dest) 32 KB
    __shared__ __bf16 Vt[HD * BLK];        // V^T [hd][key]               32 KB
    __shared__ __bf16 Pt[8 * 16 * BLK];    // per-wave P tile             32 KB
    __shared__ float  km_s[BLK];           // additive key mask

    const int n = blockIdx.x, h = blockIdx.y;
    const int kv = h >> 2;                 // NH/NKV = 4
    const int tid  = threadIdx.x;
    const int wave = tid >> 5, lane = tid & 31;
    const int half = lane >> 4, lm = lane & 15;
    const uint32_t kraw_off = (uint32_t)(uintptr_t)(&Kraw[0]);

    // Preload Q stripe in A-operand layout (4 chunks of K=32 over HD=128)
    const __bf16* qrow = Q + (size_t)(n * BLK + wave * 16 + lm) * (NH * HD) + h * HD;
    v16bf qa[4];
#pragma unroll
    for (int kc = 0; kc < 4; ++kc) {
        v8bf lo = *(const v8bf*)(qrow + kc * 32 + half * 8);
        v8bf hi = *(const v8bf*)(qrow + kc * 32 + 16 + half * 8);
        qa[kc] = cat8(lo, hi);
    }

    const v8f zero = {0.f,0.f,0.f,0.f,0.f,0.f,0.f,0.f};
    float mrun[8], lrun[8];
    v8f acc[8];
#pragma unroll
    for (int r = 0; r < 8; ++r) { mrun[r] = -3.0e38f; lrun[r] = 0.f; acc[r] = zero; }

    __bf16* ptw = &Pt[wave * 16 * BLK];

    for (int kb = 0; kb < LOCAL + GLOB; ++kb) {
        // ---- key block index + validity (matches reference idx/dedup) ----
        int bn; float bvalid = 1.f;
        if (kb < LOCAL) {
            bn = n - (LOCAL - 1) + kb;
            if (bn < 0) { bn = 0; bvalid = 0.f; }
        } else {
            int g = kb - LOCAL;
            int i1 = n - 3 * KSTRIDE; if (i1 < 0) i1 = 0;
            int i2 = n - 2 * KSTRIDE; if (i2 < 0) i2 = 0;
            int i3 = n - 1 * KSTRIDE; if (i3 < 0) i3 = 0;
            int idxs[4] = { 0, i1, i2, i3 };
            bn = idxs[g];
            if (g > 0 && idxs[g] == idxs[g - 1]) bvalid = 0.f;
        }

        __syncthreads();   // previous iteration's Kraw/Vt readers done

        if (tid < 32) {
            // wave 0: TDM copy of K tile [128 keys][128 hd] into Kraw
            tdm_load_2d_bf16(kraw_off,
                             Kc + (size_t)(bn * BLK) * (NKV * HD) + kv * HD,
                             /*tensor_d0=*/NKV * HD, /*tensor_d1=*/S_LEN,
                             /*stride0=*/NKV * HD,
                             /*tile_d0=*/HD, /*tile_d1=*/BLK);
            __builtin_amdgcn_s_wait_tensorcnt(0);
        }
        if (tid < BLK) {
            float kmv = mask2d[bn * BLK + tid] * bvalid;
            km_s[tid] = (1.f - kmv) * NEGINF;
        }
        {   // stage V^T: 2 threads per key row, 64 hd each (each elem read once)
            int key = tid >> 1;
            int hb  = (tid & 1) << 6;
            const __bf16* vsrc = Vc + (size_t)(bn * BLK + key) * (NKV * HD) + kv * HD + hb;
#pragma unroll
            for (int i = 0; i < 8; ++i) {
                v8bf vv = *(const v8bf*)(vsrc + i * 8);
#pragma unroll
                for (int j = 0; j < 8; ++j) Vt[(hb + i * 8 + j) * BLK + key] = vv[j];
            }
        }
        __syncthreads();

        // ---- scores: S = Q * K^T (B operand from LDS K tile) ----
        v8f sc[8];
#pragma unroll
        for (int nt = 0; nt < 8; ++nt) {
            v8f s = zero;
            const __bf16* krow = &Kraw[(nt * 16 + lm) * HD];
#pragma unroll
            for (int kc = 0; kc < 4; ++kc) {
                v8bf blo = *(const v8bf*)(krow + kc * 32 + half * 8);
                v8bf bhi = *(const v8bf*)(krow + kc * 32 + 16 + half * 8);
                s = wmma_bf16(qa[kc], cat8(blo, bhi), s);
            }
            float am   = km_s[nt * 16 + lm];
            int   kpos = nt * 16 + lm;
#pragma unroll
            for (int r = 0; r < 8; ++r) {
                float v = s[r] * SCALING + am;
                if (kb == LOCAL - 1) {               // causal inside own block
                    int qpos = wave * 16 + r + half * 8;
                    if (kpos > qpos) v += NEGINF;
                }
                s[r] = v;
            }
            sc[nt] = s;
        }

        // ---- online softmax (rows striped as r + half*8 across 16 lanes) ----
#pragma unroll
        for (int r = 0; r < 8; ++r) {
            float m = sc[0][r];
#pragma unroll
            for (int nt = 1; nt < 8; ++nt) m = fmaxf(m, sc[nt][r]);
#pragma unroll
            for (int off = 1; off < 16; off <<= 1) m = fmaxf(m, __shfl_xor(m, off, 32));
            float mnew  = fmaxf(mrun[r], m);
            float alpha = __expf(mrun[r] - mnew);
            mrun[r] = mnew;
            lrun[r] *= alpha;
#pragma unroll
            for (int nt = 0; nt < 8; ++nt) acc[nt][r] *= alpha;
            float psum = 0.f;
#pragma unroll
            for (int nt = 0; nt < 8; ++nt) {
                float pv = __expf(sc[nt][r] - mnew);
                sc[nt][r] = pv;
                psum += pv;
            }
#pragma unroll
            for (int off = 1; off < 16; off <<= 1) psum += __shfl_xor(psum, off, 32);
            lrun[r] += psum;
        }

        // ---- P: C-layout -> LDS row-major [16][128] (per-wave region) ----
#pragma unroll
        for (int nt = 0; nt < 8; ++nt)
#pragma unroll
            for (int r = 0; r < 8; ++r)
                ptw[(r + half * 8) * BLK + nt * 16 + lm] = f2bf(sc[nt][r]);

        // ---- O += P * V ----
#pragma unroll
        for (int kc = 0; kc < 4; ++kc) {
            v8bf plo = *(const v8bf*)(ptw + lm * BLK + kc * 32 + half * 8);
            v8bf phi = *(const v8bf*)(ptw + lm * BLK + kc * 32 + 16 + half * 8);
            v16bf pa = cat8(plo, phi);
#pragma unroll
            for (int nt = 0; nt < 8; ++nt) {
                const __bf16* vtp = &Vt[(nt * 16 + lm) * BLK + kc * 32];
                v8bf blo = *(const v8bf*)(vtp + half * 8);
                v8bf bhi = *(const v8bf*)(vtp + 16 + half * 8);
                acc[nt] = wmma_bf16(pa, cat8(blo, bhi), acc[nt]);
            }
        }
    }

    // ---- normalize + store attn output (bf16, [S][NH*HD]) ----
#pragma unroll
    for (int r = 0; r < 8; ++r) {
        float inv = 1.f / lrun[r];
        int row = n * BLK + wave * 16 + r + half * 8;
#pragma unroll
        for (int nt = 0; nt < 8; ++nt)
            O[(size_t)row * (NH * HD) + h * HD + nt * 16 + lm] = f2bf(acc[nt][r] * inv);
    }
}

// ---------------------------------------------------------------------------
// Host-side launch
// ---------------------------------------------------------------------------
extern "C" void kernel_launch(void* const* d_in, const int* in_sizes, int n_in,
                              void* d_out, int out_size, void* d_ws, size_t ws_size,
                              hipStream_t stream) {
    const float* hs   = (const float*)d_in[0];
    const float* mask = (const float*)d_in[1];
    const float* cosb = (const float*)d_in[2];
    const float* sinb = (const float*)d_in[3];
    const float* Wq   = (const float*)d_in[4];
    const float* Wk   = (const float*)d_in[5];
    const float* Wv   = (const float*)d_in[6];
    const float* Wo   = (const float*)d_in[7];
    float* out = (float*)d_out;

    char* p = (char*)d_ws;
    auto alloc = [&](size_t bytes) -> char* {
        char* r = p; p += (bytes + 255) & ~(size_t)255; return r;
    };
    __bf16* Hb  = (__bf16*)alloc((size_t)S_LEN * HIDDEN * 2);
    __bf16* Wqb = (__bf16*)alloc((size_t)HIDDEN * (NH * HD) * 2);
    __bf16* Wkb = (__bf16*)alloc((size_t)HIDDEN * (NKV * HD) * 2);
    __bf16* Wvb = (__bf16*)alloc((size_t)HIDDEN * (NKV * HD) * 2);
    __bf16* Wob = (__bf16*)alloc((size_t)(NH * HD) * HIDDEN * 2);
    __bf16* Qb  = (__bf16*)alloc((size_t)S_LEN * (NH * HD) * 2);
    __bf16* Kb  = (__bf16*)alloc((size_t)S_LEN * (NKV * HD) * 2);
    __bf16* Vb  = (__bf16*)alloc((size_t)S_LEN * (NKV * HD) * 2);
    __bf16* AOb = (__bf16*)alloc((size_t)S_LEN * (NH * HD) * 2);
    (void)in_sizes; (void)n_in; (void)out_size; (void)ws_size;

    auto cvt = [&](const float* s, __bf16* d, int n) {
        cvt_f32_bf16<<<(n + 255) / 256, 256, 0, stream>>>(s, d, n);
    };
    cvt(hs, Hb, S_LEN * HIDDEN);
    cvt(Wq, Wqb, HIDDEN * NH * HD);
    cvt(Wk, Wkb, HIDDEN * NKV * HD);
    cvt(Wv, Wvb, HIDDEN * NKV * HD);
    cvt(Wo, Wob, (NH * HD) * HIDDEN);

    // QKV projections
    gemm_bf16_kernel<true><<<dim3((NH * HD) / 128, S_LEN / 128), 256, 0, stream>>>(
        Hb, Wqb, (void*)Qb, S_LEN, NH * HD, HIDDEN);
    gemm_bf16_kernel<true><<<dim3((NKV * HD) / 128, S_LEN / 128), 256, 0, stream>>>(
        Hb, Wkb, (void*)Kb, S_LEN, NKV * HD, HIDDEN);
    gemm_bf16_kernel<true><<<dim3((NKV * HD) / 128, S_LEN / 128), 256, 0, stream>>>(
        Hb, Wvb, (void*)Vb, S_LEN, NKV * HD, HIDDEN);

    // RoPE on Q and K
    {
        int nq = S_LEN * NH * HD / 2;
        rope_kernel<<<(nq + 255) / 256, 256, 0, stream>>>(Qb, cosb, sinb, NH * HD, nq);
        int nk = S_LEN * NKV * HD / 2;
        rope_kernel<<<(nk + 255) / 256, 256, 0, stream>>>(Kb, cosb, sinb, NKV * HD, nk);
    }

    // Attention
    attn_kernel<<<dim3(NBLK, NH), 256, 0, stream>>>(Qb, Kb, Vb, mask, AOb);

    // Output projection (fp32 out)
    gemm_bf16_kernel<false><<<dim3(HIDDEN / 128, S_LEN / 128), 256, 0, stream>>>(
        AOb, Wob, (void*)out, S_LEN, HIDDEN, HIDDEN);
}
